// Decoder_17162689315582
// MI455X (gfx1250) — matrix operand order repaired and checked by reading, not compiled, not run on hardware
//
#include <hip/hip_runtime.h>

// ---------------- constants ----------------
#define NWG 128
#define TPB 256
#define BB  64
#define SS  256
#define TT  800
#define NG  4096
#define KA  1792   // [x(256) | ctx(512) | h_a(1024)]
#define KD  2560   // [h_a(1024) | ctx(512) | h_d(1024)]
#define KP  1536   // [h_d(1024) | ctx(512)]

typedef __attribute__((ext_vector_type(16))) unsigned short us16;
typedef __attribute__((ext_vector_type(16))) __bf16         v16bf;
typedef __attribute__((ext_vector_type(8)))  float          v8f;

// ---------------- bf16 helpers ----------------
__device__ __forceinline__ unsigned short f2bf(float f) {
  unsigned int u = __float_as_uint(f);
  u += 0x7fffu + ((u >> 16) & 1u);          // round-to-nearest-even
  return (unsigned short)(u >> 16);
}
__device__ __forceinline__ float bf2f(unsigned short h) {
  return __uint_as_float(((unsigned int)h) << 16);
}
__device__ __forceinline__ float sigf(float x) { return 1.f / (1.f + __expf(-x)); }

// pack two 8-half contiguous chunks into one 16-elem bf16 fragment
__device__ __forceinline__ us16 packh(const unsigned short* p0, const unsigned short* p1) {
  union U { uint4 q; unsigned short s[8]; };
  U a; a.q = *(const uint4*)p0;
  U b; b.q = *(const uint4*)p1;
  us16 r;
#pragma unroll
  for (int i = 0; i < 8; ++i) { r[i] = a.s[i]; r[i + 8] = b.s[i]; }
  return r;
}
// same but converting from f32 on the fly
__device__ __forceinline__ us16 packf(const float* p0, const float* p1) {
  us16 r;
#pragma unroll
  for (int i = 0; i < 8; ++i) { r[i] = f2bf(p0[i]); r[i + 8] = f2bf(p1[i]); }
  return r;
}
__device__ __forceinline__ v8f wmma_bf(us16 a, us16 b, v8f c) {
  return __builtin_amdgcn_wmma_f32_16x16x32_bf16(
      false, __builtin_bit_cast(v16bf, a),
      false, __builtin_bit_cast(v16bf, b),
      (short)0, c, false, false);
}

// ---------------- workspace layout (bytes) ----------------
constexpr size_t AL(size_t x) { return (x + 255) & ~(size_t)255; }
constexpr size_t OFF_WA  = 0;                                        // ushort 4096*1792
constexpr size_t OFF_WD  = OFF_WA  + AL((size_t)NG * KA * 2);        // ushort 4096*2560
constexpr size_t OFF_WP  = OFF_WD  + AL((size_t)NG * KD * 2);        // ushort 96*1536
constexpr size_t OFF_BP  = OFF_WP  + AL((size_t)96 * KP * 2);        // float 96
constexpr size_t OFF_X2  = OFF_BP  + AL(96 * 4);                     // ushort 51200*256
constexpr size_t OFF_PM  = OFF_X2  + AL((size_t)TT * BB * 256 * 2);  // float 16384*128
constexpr size_t OFF_PL  = OFF_PM  + AL((size_t)BB * SS * 128 * 4);  // float 16384*128
constexpr size_t OFF_GA  = OFF_PL  + AL((size_t)BB * SS * 128 * 4);  // float 64*4096
constexpr size_t OFF_GD  = OFF_GA  + AL((size_t)BB * NG * 4);        // float 64*4096
constexpr size_t OFF_HA  = OFF_GD  + AL((size_t)BB * NG * 4);        // ushort 64*1024
constexpr size_t OFF_HD  = OFF_HA  + AL((size_t)BB * 1024 * 2);      // ushort 64*1024
constexpr size_t OFF_CTX = OFF_HD  + AL((size_t)BB * 1024 * 2);      // ushort 64*512
constexpr size_t OFF_CA  = OFF_CTX + AL((size_t)BB * 512 * 2);       // float 64*1024
constexpr size_t OFF_CD  = OFF_CA  + AL((size_t)BB * 1024 * 4);      // float 64*1024
constexpr size_t OFF_AW  = OFF_CD  + AL((size_t)BB * 1024 * 4);      // float 64*256
constexpr size_t OFF_AWC = OFF_AW  + AL((size_t)BB * SS * 4);        // float 64*256
constexpr size_t OFF_CNT = OFF_AWC + AL((size_t)BB * SS * 4);        // uint
constexpr size_t WS_NEED = OFF_CNT + 256;

struct P {
  const float* memory; const int* memlen;
  const float *abih, *abhh, *dbih, *dbhh;
  const float *queryw, *vw, *convw, *densew;
  unsigned short *Wa, *Wd, *Wp, *X2, *HA, *HD, *CTX;
  float *BP, *PM, *PL, *GA, *GD, *CA, *CD, *AW, *AWC;
  unsigned int* cnt;
  float *out_mel, *out_gate, *out_align;
};

// ---------------- init / weight conversion ----------------
__global__ void k_init(char* ws, const float* wihA, const float* whhA,
                       const float* wihD, const float* whhD,
                       const float* projw, const float* projb,
                       const float* gatew, const float* gateb) {
  size_t t0 = (size_t)blockIdx.x * blockDim.x + threadIdx.x;
  size_t st = (size_t)gridDim.x * blockDim.x;
  unsigned short* Wa = (unsigned short*)(ws + OFF_WA);
  unsigned short* Wd = (unsigned short*)(ws + OFF_WD);
  unsigned short* Wp = (unsigned short*)(ws + OFF_WP);
  float* BP = (float*)(ws + OFF_BP);
  unsigned short* HA = (unsigned short*)(ws + OFF_HA);
  unsigned short* HD = (unsigned short*)(ws + OFF_HD);
  unsigned short* CTX = (unsigned short*)(ws + OFF_CTX);
  float* CA = (float*)(ws + OFF_CA);
  float* CD = (float*)(ws + OFF_CD);
  float* AW = (float*)(ws + OFF_AW);
  float* AWC = (float*)(ws + OFF_AWC);
  for (size_t i = t0; i < (size_t)NG * KA; i += st) {
    size_t n = i / KA, k = i % KA;
    float v = (k < 768) ? wihA[n * 768 + k] : whhA[n * 1024 + (k - 768)];
    Wa[i] = f2bf(v);
  }
  for (size_t i = t0; i < (size_t)NG * KD; i += st) {
    size_t n = i / KD, k = i % KD;
    float v = (k < 1536) ? wihD[n * 1536 + k] : whhD[n * 1024 + (k - 1536)];
    Wd[i] = f2bf(v);
  }
  for (size_t i = t0; i < (size_t)96 * KP; i += st) {
    size_t n = i / KP, k = i % KP;
    float v = (n < 80) ? projw[n * KP + k] : ((n == 80) ? gatew[k] : 0.f);
    Wp[i] = f2bf(v);
  }
  for (size_t i = t0; i < 96; i += st)
    BP[i] = (i < 80) ? projb[i] : ((i == 80) ? gateb[0] : 0.f);
  for (size_t i = t0; i < (size_t)BB * 1024; i += st) {
    HA[i] = 0; HD[i] = 0; CA[i] = 0.f; CD[i] = 0.f;
  }
  for (size_t i = t0; i < (size_t)BB * 512; i += st) CTX[i] = 0;
  for (size_t i = t0; i < (size_t)BB * SS; i += st) { AW[i] = 0.f; AWC[i] = 0.f; }
  if (t0 == 0) *(unsigned int*)(ws + OFF_CNT) = 0u;
}

// ---------------- fused prenet (two Linear+ReLU), bf16 out ----------------
__global__ void __launch_bounds__(TPB) k_prenet(char* ws, const float* dec_in,
                                                const float* w1, const float* w2) {
  __shared__ unsigned short w2t[256 * 256];  // transposed: [k][j]
  __shared__ float xh[8][80 + 256];
  for (int i = threadIdx.x; i < 256 * 256; i += TPB) {
    int j = i >> 8, k = i & 255;
    w2t[k * 256 + j] = f2bf(w2[i]);          // w2[j][k] -> w2t[k][j]
  }
  __syncthreads();
  unsigned short* X2 = (unsigned short*)(ws + OFF_X2);
  int wave = threadIdx.x >> 5, lane = threadIdx.x & 31;
  float* xs = xh[wave];
  float* h1 = xh[wave] + 80;
  for (int row = blockIdx.x * 8 + wave; row < TT * BB; row += NWG * 8) {
    int t = row >> 6, b = row & 63;          // row = t*64 + b
    for (int k = lane; k < 80; k += 32)
      xs[k] = (t == 0) ? 0.f : dec_in[((size_t)b * 80 + k) * TT + (t - 1)];
#pragma unroll
    for (int jj = 0; jj < 8; ++jj) {
      int j = lane + jj * 32; float acc = 0.f;
      const float* wr = w1 + j * 80;
      for (int k = 0; k < 80; ++k) acc += xs[k] * wr[k];
      h1[j] = fmaxf(acc, 0.f);
    }
#pragma unroll
    for (int jj = 0; jj < 8; ++jj) {
      int j = lane + jj * 32; float acc = 0.f;
      for (int k = 0; k < 256; ++k) acc += h1[k] * bf2f(w2t[k * 256 + j]);
      X2[(size_t)row * 256 + j] = f2bf(fmaxf(acc, 0.f));
    }
  }
}

// ---------------- processed_memory = memory @ memory_w^T (WMMA) ----------------
__global__ void __launch_bounds__(TPB) k_procmem(char* ws, const float* memory,
                                                 const float* memw) {
  float* PM = (float*)(ws + OFF_PM);
  int lane = threadIdx.x & 31, wave = threadIdx.x >> 5;
  int hi = lane >> 4, lm = lane & 15;
  for (int item = blockIdx.x * 8 + wave; item < 1024 * 8; item += gridDim.x * 8) {
    int mt = item >> 3, nt = item & 7;
    int arow = mt * 16 + lm;                 // = b*256+s
    int bcol = nt * 16 + lm;
    v8f acc = {};
    for (int kb = 0; kb < 512; kb += 32) {
      const float* pa = memory + (size_t)arow * 512 + kb + hi * 8;
      const float* pb = memw + (size_t)bcol * 512 + kb + hi * 8;
      acc = wmma_bf(packf(pa, pa + 16), packf(pb, pb + 16), acc);
    }
#pragma unroll
    for (int r = 0; r < 8; ++r)
      PM[(size_t)(mt * 16 + hi * 8 + r) * 128 + nt * 16 + lm] = acc[r];
  }
}

// ---------------- persistent decoder ----------------
__device__ __forceinline__ void gbar(unsigned int* cnt, unsigned int* gen) {
  __syncthreads();
  if (threadIdx.x == 0) {
    __threadfence();
    unsigned int target = (++(*gen)) * (unsigned int)NWG;
    atomicAdd(cnt, 1u);
    while (__hip_atomic_load(cnt, __ATOMIC_ACQUIRE, __HIP_MEMORY_SCOPE_AGENT) < target)
      __builtin_amdgcn_s_sleep(1);
  }
  __syncthreads();
}

__device__ void gemmA_tile(const P& p, int item, int t, int lane) {
  int mt = item >> 8, nt = item & 255;
  int hi = lane >> 4, lm = lane & 15;
  int arow = mt * 16 + lm, bcol = nt * 16 + lm;
  const unsigned short* Wrow = p.Wa + (size_t)bcol * KA;
  v8f acc = {};
  for (int kb = 0; kb < KA; kb += 32) {
    const unsigned short* pa;
    if (kb < 256)      pa = p.X2 + ((size_t)t * 64 + arow) * 256 + kb;
    else if (kb < 768) pa = p.CTX + (size_t)arow * 512 + (kb - 256);
    else               pa = p.HA + (size_t)arow * 1024 + (kb - 768);
    pa += hi * 8;
    const unsigned short* pb = Wrow + kb + hi * 8;
    acc = wmma_bf(packh(pa, pa + 16), packh(pb, pb + 16), acc);
  }
#pragma unroll
  for (int r = 0; r < 8; ++r)
    p.GA[(size_t)(mt * 16 + hi * 8 + r) * NG + nt * 16 + lm] = acc[r];
}

__device__ void gemmD_tile(const P& p, int item, int lane) {
  int mt = item >> 8, nt = item & 255;
  int hi = lane >> 4, lm = lane & 15;
  int arow = mt * 16 + lm, bcol = nt * 16 + lm;
  const unsigned short* Wrow = p.Wd + (size_t)bcol * KD;
  v8f acc = {};
  for (int kb = 0; kb < KD; kb += 32) {
    const unsigned short* pa;
    if (kb < 1024)      pa = p.HA + (size_t)arow * 1024 + kb;
    else if (kb < 1536) pa = p.CTX + (size_t)arow * 512 + (kb - 1024);
    else                pa = p.HD + (size_t)arow * 1024 + (kb - 1536);
    pa += hi * 8;
    const unsigned short* pb = Wrow + kb + hi * 8;
    acc = wmma_bf(packh(pa, pa + 16), packh(pb, pb + 16), acc);
  }
#pragma unroll
  for (int r = 0; r < 8; ++r)
    p.GD[(size_t)(mt * 16 + hi * 8 + r) * NG + nt * 16 + lm] = acc[r];
}

__device__ void proj_tile(const P& p, int item, int tt, int lane) {
  int mt = item / 6, nt = item % 6;
  int hi = lane >> 4, lm = lane & 15;
  int arow = mt * 16 + lm, bcol = nt * 16 + lm;
  const unsigned short* Wrow = p.Wp + (size_t)bcol * KP;
  v8f acc = {};
  for (int kb = 0; kb < KP; kb += 32) {
    const unsigned short* pa;
    if (kb < 1024) pa = p.HD + (size_t)arow * 1024 + kb;
    else           pa = p.CTX + (size_t)arow * 512 + (kb - 1024);
    pa += hi * 8;
    const unsigned short* pb = Wrow + kb + hi * 8;
    acc = wmma_bf(packh(pa, pa + 16), packh(pb, pb + 16), acc);
  }
#pragma unroll
  for (int r = 0; r < 8; ++r) {
    int b = mt * 16 + hi * 8 + r;
    int col = nt * 16 + lm;
    float v = acc[r] + p.BP[col];
    if (col < 80)        p.out_mel[((size_t)b * 80 + col) * TT + tt] = v;
    else if (col == 80)  p.out_gate[(size_t)b * TT + tt] = v;
  }
}

__device__ void conv_item(const P& p, int item, int lane, float* convbuf) {
  int b = item >> 4, st = item & 15, s0 = st << 4;
  const float* wrow = p.convw + lane * 62;   // [f][2][31]; lane = filter
  const float* awb = p.AW + b * SS;
  const float* awcb = p.AWC + b * SS;
  for (int ss = 0; ss < 16; ++ss) {
    int s = s0 + ss; float acc = 0.f;
#pragma unroll
    for (int j = 0; j < 31; ++j) {
      int sp = s + j - 15;
      if (sp >= 0 && sp < SS) acc += wrow[j] * awb[sp] + wrow[31 + j] * awcb[sp];
    }
    convbuf[ss * 32 + lane] = acc;           // in-order LDS within wave
  }
  for (int o = lane; o < 16 * 128; o += 32) {
    int ss = o >> 7, a = o & 127;
    float acc = 0.f;
    const float* dw = p.densew + a * 32;
#pragma unroll
    for (int f = 0; f < 32; ++f) acc += convbuf[ss * 32 + f] * dw[f];
    p.PL[((size_t)(b * SS + s0 + ss)) * 128 + a] = acc;
  }
}

__device__ void attention(const P& p, int b, int t, int tid, float* smem) {
  float* ha  = smem;            // 1024
  float* q   = smem + 1024;     // 128
  float* awl = smem + 1152;     // 256
  float* red = smem + 1408;     // 256
  // 1) attn-LSTM pointwise
  for (int j = tid; j < 1024; j += TPB) {
    float ig = p.GA[(size_t)b * NG + j]        + p.abih[j]        + p.abhh[j];
    float fg = p.GA[(size_t)b * NG + 1024 + j] + p.abih[1024 + j] + p.abhh[1024 + j];
    float gg = p.GA[(size_t)b * NG + 2048 + j] + p.abih[2048 + j] + p.abhh[2048 + j];
    float og = p.GA[(size_t)b * NG + 3072 + j] + p.abih[3072 + j] + p.abhh[3072 + j];
    float c = sigf(fg) * p.CA[b * 1024 + j] + sigf(ig) * tanhf(gg);
    float h = sigf(og) * tanhf(c);
    p.CA[b * 1024 + j] = c;
    ha[j] = h;
    p.HA[b * 1024 + j] = f2bf(h);
  }
  __syncthreads();
  // 2) q = h_a @ query_w^T
  if (tid < 128) {
    float acc = 0.f;
    const float* wr = p.queryw + tid * 1024;
    for (int k = 0; k < 1024; ++k) acc += ha[k] * wr[k];
    q[tid] = acc;
  }
  __syncthreads();
  // 3) energies + masked softmax  (TPB == SS)
  int s = tid;
  float e;
  {
    const float* pl = p.PL + ((size_t)(b * SS + s)) * 128;
    const float* pm = p.PM + ((size_t)(b * SS + s)) * 128;
    float acc = 0.f;
    for (int a = 0; a < 128; ++a) acc += p.vw[a] * tanhf(q[a] + pl[a] + pm[a]);
    e = (s >= p.memlen[b]) ? -1e9f : acc;
  }
  red[tid] = e; __syncthreads();
  for (int off = 128; off; off >>= 1) {
    if (tid < off) red[tid] = fmaxf(red[tid], red[tid + off]);
    __syncthreads();
  }
  float m = red[0]; __syncthreads();
  float ex = __expf(e - m);
  red[tid] = ex; __syncthreads();
  for (int off = 128; off; off >>= 1) {
    if (tid < off) red[tid] += red[tid + off];
    __syncthreads();
  }
  float aw = ex / red[0];
  awl[s] = aw;
  p.AW[b * SS + s] = aw;
  p.AWC[b * SS + s] += aw;
  p.out_align[((size_t)b * TT + t) * SS + s] = aw;
  __syncthreads();
  // 4) context = aw @ memory[b]
  for (int e2 = tid; e2 < 512; e2 += TPB) {
    float acc = 0.f;
    const float* mem = p.memory + ((size_t)b * SS) * 512 + e2;
    for (int s2 = 0; s2 < SS; ++s2) acc += awl[s2] * mem[(size_t)s2 * 512];
    p.CTX[b * 512 + e2] = f2bf(acc);
  }
}

__device__ void lstmD_elem(const P& p, int i) {
  int b = i >> 10, j = i & 1023;
  float ig = p.GD[(size_t)b * NG + j]        + p.dbih[j]        + p.dbhh[j];
  float fg = p.GD[(size_t)b * NG + 1024 + j] + p.dbih[1024 + j] + p.dbhh[1024 + j];
  float gg = p.GD[(size_t)b * NG + 2048 + j] + p.dbih[2048 + j] + p.dbhh[2048 + j];
  float og = p.GD[(size_t)b * NG + 3072 + j] + p.dbih[3072 + j] + p.dbhh[3072 + j];
  float c = sigf(fg) * p.CD[i] + sigf(ig) * tanhf(gg);
  float h = sigf(og) * tanhf(c);
  p.CD[i] = c;
  p.HD[i] = f2bf(h);
}

__global__ void __launch_bounds__(TPB) k_decoder(P p) {
  __shared__ float smem[4096];
  const int tid = threadIdx.x, lane = tid & 31, wave = tid >> 5;
  const int wg = blockIdx.x;
  unsigned int gen = 0;
  for (int t = 0; t < TT; ++t) {
    // Phase A: attn-LSTM gate GEMM + location conv/dense + projection(t-1)
    int total = 2048 + ((t > 0) ? 24 : 0);
    for (int item = wg * 8 + wave; item < total; item += NWG * 8) {
      if (item < 1024)      gemmA_tile(p, item, t, lane);
      else if (item < 2048) conv_item(p, item - 1024, lane, smem + wave * 512);
      else                  proj_tile(p, item - 2048, t - 1, lane);
    }
    gbar(p.cnt, &gen);
    // Phase B: per-batch attention (LSTM pointwise, q, energies, softmax, ctx)
    if (wg < BB) attention(p, wg, t, tid, smem);
    gbar(p.cnt, &gen);
    // Phase C: dec-LSTM gate GEMM
    for (int item = wg * 8 + wave; item < 1024; item += NWG * 8)
      gemmD_tile(p, item, lane);
    gbar(p.cnt, &gen);
    // Phase D: dec-LSTM pointwise
    for (int i = wg * TPB + tid; i < BB * 1024; i += NWG * TPB)
      lstmD_elem(p, i);
    gbar(p.cnt, &gen);
  }
  // epilogue: projection for final step
  for (int item = wg * 8 + wave; item < 24; item += NWG * 8)
    proj_tile(p, item, TT - 1, lane);
}

// ---------------- host launch ----------------
extern "C" void kernel_launch(void* const* d_in, const int* in_sizes, int n_in,
                              void* d_out, int out_size, void* d_ws, size_t ws_size,
                              hipStream_t stream) {
  (void)in_sizes; (void)n_in; (void)out_size;
  if (ws_size < WS_NEED) return;
  const float* memory = (const float*)d_in[0];
  const float* dec_in = (const float*)d_in[1];
  const int*   memlen = (const int*)d_in[2];
  const float* w1     = (const float*)d_in[3];
  const float* w2     = (const float*)d_in[4];
  const float* awih   = (const float*)d_in[5];
  const float* awhh   = (const float*)d_in[6];
  const float* abih   = (const float*)d_in[7];
  const float* abhh   = (const float*)d_in[8];
  const float* dwih   = (const float*)d_in[9];
  const float* dwhh   = (const float*)d_in[10];
  const float* dbih   = (const float*)d_in[11];
  const float* dbhh   = (const float*)d_in[12];
  const float* qw     = (const float*)d_in[13];
  const float* mw     = (const float*)d_in[14];
  const float* vw     = (const float*)d_in[15];
  const float* convw  = (const float*)d_in[16];
  const float* densew = (const float*)d_in[17];
  const float* projw  = (const float*)d_in[18];
  const float* projb  = (const float*)d_in[19];
  const float* gatew  = (const float*)d_in[20];
  const float* gateb  = (const float*)d_in[21];
  char* ws = (char*)d_ws;

  k_init<<<512, 256, 0, stream>>>(ws, awih, awhh, dwih, dwhh, projw, projb, gatew, gateb);
  k_prenet<<<NWG, TPB, 0, stream>>>(ws, dec_in, w1, w2);
  k_procmem<<<256, TPB, 0, stream>>>(ws, memory, mw);

  P p;
  p.memory = memory; p.memlen = memlen;
  p.abih = abih; p.abhh = abhh; p.dbih = dbih; p.dbhh = dbhh;
  p.queryw = qw; p.vw = vw; p.convw = convw; p.densew = densew;
  p.Wa  = (unsigned short*)(ws + OFF_WA);
  p.Wd  = (unsigned short*)(ws + OFF_WD);
  p.Wp  = (unsigned short*)(ws + OFF_WP);
  p.X2  = (unsigned short*)(ws + OFF_X2);
  p.HA  = (unsigned short*)(ws + OFF_HA);
  p.HD  = (unsigned short*)(ws + OFF_HD);
  p.CTX = (unsigned short*)(ws + OFF_CTX);
  p.BP  = (float*)(ws + OFF_BP);
  p.PM  = (float*)(ws + OFF_PM);
  p.PL  = (float*)(ws + OFF_PL);
  p.GA  = (float*)(ws + OFF_GA);
  p.GD  = (float*)(ws + OFF_GD);
  p.CA  = (float*)(ws + OFF_CA);
  p.CD  = (float*)(ws + OFF_CD);
  p.AW  = (float*)(ws + OFF_AW);
  p.AWC = (float*)(ws + OFF_AWC);
  p.cnt = (unsigned int*)(ws + OFF_CNT);
  p.out_mel   = (float*)d_out;
  p.out_gate  = (float*)d_out + (size_t)BB * 80 * TT;
  p.out_align = (float*)d_out + (size_t)BB * 80 * TT + (size_t)BB * TT;

  k_decoder<<<NWG, TPB, 0, stream>>>(p);
}
